// ExampleModel_4209067950430
// MI455X (gfx1250) — matrix operand order repaired and checked by reading, not compiled, not run on hardware
//
#include <hip/hip_runtime.h>

// ---------------- Problem constants ----------------
#define BB 8
#define NN 4096
#define TT 32768      // B*N tokens
#define DD 1024       // model dim (= GEMM K)
#define HH 1024       // hidden dim (= GEMM N)
#define EE 16         // experts
#define CC 2048       // capacity per expert (T/E)

typedef __attribute__((ext_vector_type(16))) __bf16          v16bf;
typedef __attribute__((ext_vector_type(8)))  __bf16          v8bf;
typedef __attribute__((ext_vector_type(8)))  float           v8f;
typedef __attribute__((ext_vector_type(8)))  unsigned short  u16x8;
typedef __attribute__((ext_vector_type(4)))  unsigned short  u16x4;

__device__ __forceinline__ unsigned short f2bf(float f) {
    union { float f; unsigned u; } v; v.f = f;
    unsigned r = v.u + 0x7FFFu + ((v.u >> 16) & 1u);   // round-to-nearest-even
    return (unsigned short)(r >> 16);
}

// ---------------- 1) Gating: logits, softmax, argmax, gate ----------------
// one wave (32 lanes) per token
__global__ void gate_kernel(const float* __restrict__ x, const float* __restrict__ wg,
                            int* __restrict__ idx, float* __restrict__ gate) {
    int wave = (blockIdx.x * blockDim.x + threadIdx.x) >> 5;
    int lane = threadIdx.x & 31;
    const float* xr = x + (size_t)wave * DD;
    float acc[EE];
#pragma unroll
    for (int e = 0; e < EE; ++e) acc[e] = 0.f;
    for (int k = lane; k < DD; k += 32) {
        float xv = xr[k];
        const float4* w4 = (const float4*)(wg + (size_t)k * EE);
#pragma unroll
        for (int q = 0; q < 4; ++q) {
            float4 w = w4[q];
            acc[4 * q + 0] += xv * w.x;
            acc[4 * q + 1] += xv * w.y;
            acc[4 * q + 2] += xv * w.z;
            acc[4 * q + 3] += xv * w.w;
        }
    }
#pragma unroll
    for (int e = 0; e < EE; ++e)
#pragma unroll
        for (int m = 16; m >= 1; m >>= 1) acc[e] += __shfl_xor(acc[e], m, 32);

    if (lane == 0) {
        float mx = acc[0]; int bi = 0;
#pragma unroll
        for (int e = 1; e < EE; ++e) if (acc[e] > mx) { mx = acc[e]; bi = e; }
        float z = 0.f;
#pragma unroll
        for (int e = 0; e < EE; ++e) z += __expf(acc[e] - mx);
        idx[wave]  = bi;
        gate[wave] = 1.0f / z;   // softmax prob of the argmax
    }
}

// ---------------- 2) Sequential slot assignment (prefix count) ----------------
#define SCAN_THREADS 512
#define TOK_PER_THR  (TT / SCAN_THREADS)   // 64
__global__ void scan_kernel(const int* __restrict__ idx, int* __restrict__ token_slot,
                            int* __restrict__ slot_token, int* __restrict__ count) {
    __shared__ int sc[EE][SCAN_THREADS];
    int tid = threadIdx.x;
    int cnt[EE];
#pragma unroll
    for (int e = 0; e < EE; ++e) cnt[e] = 0;
    int t0 = tid * TOK_PER_THR;
    for (int k = 0; k < TOK_PER_THR; ++k) cnt[idx[t0 + k]]++;
#pragma unroll
    for (int e = 0; e < EE; ++e) sc[e][tid] = cnt[e];
    __syncthreads();
    for (int s = 1; s < SCAN_THREADS; s <<= 1) {
        int v[EE];
#pragma unroll
        for (int e = 0; e < EE; ++e) v[e] = (tid >= s) ? sc[e][tid - s] : 0;
        __syncthreads();
#pragma unroll
        for (int e = 0; e < EE; ++e) sc[e][tid] += v[e];
        __syncthreads();
    }
    int off[EE];
#pragma unroll
    for (int e = 0; e < EE; ++e) off[e] = sc[e][tid] - cnt[e];   // exclusive
    if (tid < EE) count[tid] = sc[tid][SCAN_THREADS - 1];
    for (int k = 0; k < TOK_PER_THR; ++k) {
        int t = t0 + k;
        int e = idx[t];
        int p = off[e]++;
        if (p < CC) {
            int s = e * CC + p;
            token_slot[t] = s;
            slot_token[s] = t;
        } else {
            token_slot[t] = -1;   // capacity drop
        }
    }
}

// ---------------- 3a) w1 [E,D,H] -> bf16 TRANSPOSED w1t [E,H,D] ----------------
// LDS-tiled 32x32 transpose: coalesced reads (h contiguous) and writes (d contiguous)
__global__ void conv_w1t_kernel(const float* __restrict__ w1, unsigned short* __restrict__ w1t) {
    __shared__ float tile[32][33];
    int e  = blockIdx.z;
    int d0 = blockIdx.y * 32;
    int h0 = blockIdx.x * 32;
    int tx = threadIdx.x & 31, ty = threadIdx.x >> 5;  // 8 rows of 32
    const float* src = w1 + (size_t)e * DD * HH;
#pragma unroll
    for (int r = 0; r < 4; ++r) {
        int d = d0 + ty + r * 8;
        tile[ty + r * 8][tx] = src[(size_t)d * HH + h0 + tx];
    }
    __syncthreads();
    unsigned short* dst = w1t + (size_t)e * HH * DD;
#pragma unroll
    for (int r = 0; r < 4; ++r) {
        int h = h0 + ty + r * 8;
        dst[(size_t)h * DD + d0 + tx] = f2bf(tile[tx][ty + r * 8]);
    }
}

// ---------------- 3b) w2 row-sums, b2 sum ----------------
__global__ void w2sum_kernel(const float* __restrict__ w2, float* __restrict__ w2s) {
    int row  = (blockIdx.x * blockDim.x + threadIdx.x) >> 5;   // e*H + h
    int lane = threadIdx.x & 31;
    const float* r = w2 + (size_t)row * DD;
    float a = 0.f;
    for (int k = lane; k < DD; k += 32) a += r[k];
#pragma unroll
    for (int m = 16; m >= 1; m >>= 1) a += __shfl_xor(a, m, 32);
    if (lane == 0) w2s[row] = a;
}

__global__ void b2sum_kernel(const float* __restrict__ b2, float* __restrict__ b2s) {
    int w = threadIdx.x >> 5, lane = threadIdx.x & 31;   // 16 waves, one per expert
    const float* r = b2 + (size_t)w * DD;
    float a = 0.f;
    for (int k = lane; k < DD; k += 32) a += r[k];
#pragma unroll
    for (int m = 16; m >= 1; m >>= 1) a += __shfl_xor(a, m, 32);
    if (lane == 0) b2s[w] = a;
}

// ---------------- 4) Dispatch: gather tokens into bf16 [E*C, D] ----------------
__global__ void dispatch_kernel(const float* __restrict__ x, const int* __restrict__ slot_token,
                                const int* __restrict__ count, unsigned short* __restrict__ bufA) {
    int s  = blockIdx.x;          // slot 0..E*C-1
    int e  = s >> 11;             // /CC
    int p  = s & (CC - 1);
    int d0 = threadIdx.x * 4;
    u16x4 o;
    if (p < count[e]) {
        int t = slot_token[s];
        float4 v = *(const float4*)(x + (size_t)t * DD + d0);
        o[0] = f2bf(v.x); o[1] = f2bf(v.y); o[2] = f2bf(v.z); o[3] = f2bf(v.w);
    } else {
        o = 0;
    }
    *(u16x4*)(bufA + (size_t)s * DD + d0) = o;
}

// ---------------- 5) WMMA GEMM + fused relu + dot(w2sum) epilogue ----------------
// grid: (nTiles=8, mTiles=16, E=16), block 256 (8 waves)
// block tile 128(M) x 128(N), K-step 32; wave computes 64x32 via 4x2 wmma 16x16x32.
// A [m][k] and B(=w1t) [n][k] tiles are both k-contiguous, staged into
// double-buffered LDS via GLOBAL_LOAD_ASYNC_TO_LDS_B128 (ASYNCcnt pipeline).
#define LDS_STRIDE 40   // ushorts per row (32 data + 8 pad), row byte stride 80 (16B aligned)
#define TILE_USHORTS (128 * LDS_STRIDE)

__device__ __forceinline__ void async_b128(unsigned lds_addr, unsigned voff,
                                           const unsigned short* sbase) {
    asm volatile("global_load_async_to_lds_b128 %0, %1, %2"
                 :: "v"(lds_addr), "v"(voff), "s"(sbase)
                 : "memory");
}

__global__ __launch_bounds__(256) void moe_gemm_kernel(
        const unsigned short* __restrict__ bufA, const unsigned short* __restrict__ w1t,
        const float* __restrict__ b1, const float* __restrict__ w2s,
        float* __restrict__ sp /* [32][E*C] partials */) {
    int nt = blockIdx.x;          // 0..7
    int mt = blockIdx.y;          // 0..15
    int e  = blockIdx.z;          // 0..15
    int tid  = threadIdx.x;
    int wave = tid >> 5, lane = tid & 31;
    int wm = wave & 1, wn = wave >> 1;       // 2 waves over M, 4 over N
    int l16 = lane & 15, half = lane >> 4;

    __shared__ unsigned short sA[2][TILE_USHORTS];  // [m][k] double-buffered
    __shared__ unsigned short sB[2][TILE_USHORTS];  // [n][k] double-buffered

    const unsigned short* Ag = bufA + ((size_t)e * CC + (size_t)mt * 128) * DD;
    const unsigned short* Bg = w1t + ((size_t)e * HH + (size_t)nt * 128) * DD;

    unsigned ldsA = (unsigned)(size_t)(&sA[0][0]);
    unsigned ldsB = (unsigned)(size_t)(&sB[0][0]);
    int      srow = tid >> 2;           // 0..63
    unsigned scc  = (tid & 3) * 8;      // 0,8,16,24

    // bias and w2sum values for this lane's two N columns (constant over K loop)
    float b1v[2], w2v[2];
#pragma unroll
    for (int j = 0; j < 2; ++j) {
        int n = nt * 128 + wn * 32 + j * 16 + l16;
        b1v[j] = b1[e * HH + n];
        w2v[j] = w2s[e * HH + n];
    }

    v8f acc[4][2];
#pragma unroll
    for (int i = 0; i < 4; ++i)
#pragma unroll
        for (int j = 0; j < 2; ++j) acc[i][j] = 0.f;

    int ka = half * 8;   // A-fragment K sub-offset per ISA 16-bit A layout

    // issue one double-buffer stage: 2 async b128 for A + 2 for B per thread
    auto stage = [&](int buf, int k0) {
        unsigned bofs = (unsigned)buf * (TILE_USHORTS * 2);  // bytes
#pragma unroll
        for (int it = 0; it < 2; ++it) {
            int r = it * 64 + srow;
            unsigned loff = (unsigned)((r * LDS_STRIDE) * 2) + scc * 2;
            unsigned goff = (unsigned)((r * DD + k0) * 2) + scc * 2;
            async_b128(ldsA + bofs + loff, goff, Ag);
            async_b128(ldsB + bofs + loff, goff, Bg);
        }
    };

    stage(0, 0);
    for (int s = 0; s < DD / 32; ++s) {
        int b = s & 1;
        if (s < DD / 32 - 1) {
            stage(b ^ 1, (s + 1) * 32);
            asm volatile("s_wait_asynccnt 0x4" ::: "memory");  // stage s resident
        } else {
            asm volatile("s_wait_asynccnt 0x0" ::: "memory");
        }
        __syncthreads();

        // A fragments: lane<16 holds K 0-7 & 16-23, lane>=16 holds K 8-15 & 24-31
        v16bf a[4];
#pragma unroll
        for (int i = 0; i < 4; ++i) {
            int row = wm * 64 + i * 16 + l16;
            v8bf lo = *(const v8bf*)(&sA[b][row * LDS_STRIDE + ka]);
            v8bf hi = *(const v8bf*)(&sA[b][row * LDS_STRIDE + ka + 16]);
            a[i] = __builtin_shufflevector(lo, hi, 0, 1, 2, 3, 4, 5, 6, 7,
                                                   8, 9, 10, 11, 12, 13, 14, 15);
        }
        // B fragments: lane<16 holds K 0-7 & 16-23, lane>=16 holds K 8-15 & 24-31
        v16bf bf[2];
#pragma unroll
        for (int j = 0; j < 2; ++j) {
            int col = wn * 32 + j * 16 + l16;
            v8bf lo = *(const v8bf*)(&sB[b][col * LDS_STRIDE + half * 8]);
            v8bf hi = *(const v8bf*)(&sB[b][col * LDS_STRIDE + half * 8 + 16]);
            bf[j] = __builtin_shufflevector(lo, hi, 0, 1, 2, 3, 4, 5, 6, 7,
                                                    8, 9, 10, 11, 12, 13, 14, 15);
        }
#pragma unroll
        for (int i = 0; i < 4; ++i)
#pragma unroll
            for (int j = 0; j < 2; ++j)
                acc[i][j] = __builtin_amdgcn_wmma_f32_16x16x32_bf16(
                    false, a[i], false, bf[j], (short)0, acc[i][j], false, false);
        __syncthreads();   // protects buffer b from being overwritten by stage s+2
    }

    // Epilogue: relu(acc + b1) dotted with w2sum over this wave's 32 N-columns,
    // reduced across the 16 lanes of each half, stored as per-(nt,wn) partials.
#pragma unroll
    for (int i = 0; i < 4; ++i) {
        float p[8];
#pragma unroll
        for (int r = 0; r < 8; ++r) p[r] = 0.f;
#pragma unroll
        for (int j = 0; j < 2; ++j)
#pragma unroll
            for (int r = 0; r < 8; ++r) {
                float h = acc[i][j][r] + b1v[j];
                h = h > 0.f ? h : 0.f;
                p[r] += h * w2v[j];
            }
#pragma unroll
        for (int m = 8; m >= 1; m >>= 1)
#pragma unroll
            for (int r = 0; r < 8; ++r) p[r] += __shfl_xor(p[r], m, 16);
        if (l16 == 0) {
            int q = nt * 4 + wn;                                   // 0..31
            float* dst = sp + (size_t)q * (EE * CC)
                            + e * CC + mt * 128 + wm * 64 + i * 16 + half * 8;
#pragma unroll
            for (int r = 0; r < 8; ++r) dst[r] = p[r];
        }
    }
}

// ---------------- 6) Combine: gather + gate + b2sum ----------------
__global__ void combine_kernel(const int* __restrict__ token_slot, const float* __restrict__ gate,
                               const float* __restrict__ sp, const float* __restrict__ b2s,
                               float* __restrict__ outsum) {
    int t = blockIdx.x * blockDim.x + threadIdx.x;
    int s = token_slot[t];
    float v = 0.f;
    if (s >= 0) {
        float a = 0.f;
#pragma unroll
        for (int q = 0; q < 32; ++q) a += sp[(size_t)q * (EE * CC) + s];
        v = gate[t] * (a + b2s[s >> 11]);
    }
    outsum[t] = v;
}

// ---------------- 7) log_softmax over N per batch row ----------------
__global__ void lsm_kernel(const float* __restrict__ outsum, float* __restrict__ out) {
    int b = blockIdx.x, tid = threadIdx.x;
    const float* r = outsum + (size_t)b * NN;
    __shared__ float red[256];
    float mx = -3.402823466e38f;
    for (int i = tid; i < NN; i += 256) mx = fmaxf(mx, r[i]);
    red[tid] = mx; __syncthreads();
    for (int s = 128; s > 0; s >>= 1) {
        if (tid < s) red[tid] = fmaxf(red[tid], red[tid + s]);
        __syncthreads();
    }
    mx = red[0]; __syncthreads();
    float sm = 0.f;
    for (int i = tid; i < NN; i += 256) sm += __expf(r[i] - mx);
    red[tid] = sm; __syncthreads();
    for (int s = 128; s > 0; s >>= 1) {
        if (tid < s) red[tid] += red[tid + s];
        __syncthreads();
    }
    float ls = __logf(red[0]);
    for (int i = tid; i < NN; i += 256) out[(size_t)b * NN + i] = r[i] - mx - ls;
}

// ---------------- Launch ----------------
extern "C" void kernel_launch(void* const* d_in, const int* in_sizes, int n_in,
                              void* d_out, int out_size, void* d_ws, size_t ws_size,
                              hipStream_t stream) {
    (void)in_sizes; (void)n_in; (void)out_size; (void)ws_size;
    const float* x  = (const float*)d_in[0];   // [B,N,D]
    const float* wg = (const float*)d_in[1];   // [D,E]
    const float* w1 = (const float*)d_in[2];   // [E,D,H]
    const float* b1 = (const float*)d_in[3];   // [E,H]
    const float* w2 = (const float*)d_in[4];   // [E,H,D]
    const float* b2 = (const float*)d_in[5];   // [E,D]
    float* out = (float*)d_out;                // [B,N]

    char* w = (char*)d_ws;
    auto alloc = [&](size_t bytes) -> void* {
        void* p = (void*)w;
        w += (bytes + 255) & ~(size_t)255;
        return p;
    };
    int*            idx        = (int*)alloc((size_t)TT * 4);
    float*          gate       = (float*)alloc((size_t)TT * 4);
    int*            token_slot = (int*)alloc((size_t)TT * 4);
    int*            slot_token = (int*)alloc((size_t)EE * CC * 4);
    int*            count      = (int*)alloc(EE * 4);
    float*          b2s        = (float*)alloc(EE * 4);
    float*          w2s        = (float*)alloc((size_t)EE * HH * 4);
    float*          sp         = (float*)alloc((size_t)32 * EE * CC * 4);
    float*          outsum     = (float*)alloc((size_t)TT * 4);
    unsigned short* bufA       = (unsigned short*)alloc((size_t)EE * CC * DD * 2);
    unsigned short* w1t        = (unsigned short*)alloc((size_t)EE * DD * HH * 2);

    conv_w1t_kernel<<<dim3(HH / 32, DD / 32, EE), 256, 0, stream>>>(w1, w1t);
    w2sum_kernel<<<(EE * HH) / 8, 256, 0, stream>>>(w2, w2s);
    b2sum_kernel<<<1, 512, 0, stream>>>(b2, b2s);
    gate_kernel<<<TT / 8, 256, 0, stream>>>(x, wg, idx, gate);
    scan_kernel<<<1, SCAN_THREADS, 0, stream>>>(idx, token_slot, slot_token, count);
    dispatch_kernel<<<EE * CC, 256, 0, stream>>>(x, slot_token, count, bufA);
    moe_gemm_kernel<<<dim3(HH / 128, CC / 128, EE), 256, 0, stream>>>(bufA, w1t, b1, w2s, sp);
    combine_kernel<<<TT / 256, 256, 0, stream>>>(token_slot, gate, sp, b2s, outsum);
    lsm_kernel<<<BB, 256, 0, stream>>>(outsum, out);
}
